// ClusteredLinformerAttention_14791867367758
// MI455X (gfx1250) — matrix operand order repaired and checked by reading, not compiled, not run on hardware
//
#include <hip/hip_runtime.h>

// ---------------------------------------------------------------------------
// Linformer-style attention for MI455X (gfx1250).
// All heavy math via v_wmma_f32_16x16x32_bf16 (f32 accumulate).
// GEMM tiles staged with GLOBAL_LOAD_ASYNC_TO_LDS_B128 (ASYNCcnt), double
// buffered, one barrier per K-step.
// conv(3,R,'SAME') re-expressed as 3 GEMMs against Toeplitz matrices.
// conv_bias is a scalar added uniformly over the softmax axis -> no-op.
// ---------------------------------------------------------------------------

typedef __bf16 bf16;
typedef __attribute__((ext_vector_type(16))) __bf16 v16bf;
typedef __attribute__((ext_vector_type(8)))  __bf16 v8bf;
typedef __attribute__((ext_vector_type(8)))  float  v8f;

static constexpr int Bb = 4, Ss = 4096, Dd = 1024, Hh = 16, Rr = 256, DEPTH = 64;
static constexpr long NTOK = (long)Bb * Ss;   // 16384

__device__ __forceinline__ v8f wmma_bf16(v16bf a, v16bf b, v8f c) {
  // D = A(16x32) * B(32x16) + C, f32 accumulate
  return __builtin_amdgcn_wmma_f32_16x16x32_bf16(false, a, false, b, (short)0, c,
                                                 false, false);
}

// Async copy 16B global -> LDS (per lane), tracked by ASYNCcnt.
// LDS operand is the 32-bit LDS byte address (generic LDS addr truncates to it).
__device__ __forceinline__ void async_ld_b128(void* lds, const void* g) {
  unsigned l = (unsigned)(unsigned long long)(uintptr_t)lds;
  asm volatile("global_load_async_to_lds_b128 %0, %1, off"
               :: "v"(l), "v"(g) : "memory");
}
__device__ __forceinline__ void wait_async0() {
  asm volatile("s_wait_asynccnt 0x0" ::: "memory");
}

// A fragment 16x32 (MxK), row-major tile, per ISA 7.12.2 16-bit A table:
// lane<16 : M=lane, K in {h*8+0..7, 16+h*8+0..7}, h = lane>>4
__device__ __forceinline__ v16bf load_frag_a(const bf16* __restrict__ p, int ld) {
  const int lane = threadIdx.x & 31;
  const int m = lane & 15;
  const int hh = lane >> 4;
  const bf16* base = p + (long)m * ld + hh * 8;
  v16bf r;
  *reinterpret_cast<v8bf*>(&r)       = *reinterpret_cast<const v8bf*>(base);
  *(reinterpret_cast<v8bf*>(&r) + 1) = *reinterpret_cast<const v8bf*>(base + 16);
  return r;
}

// B fragment 32x16 (KxN) loaded from column-major storage Bt[N][K]:
// lane half h holds K in {h*8+0..7, 16+h*8+0..7} for column n = lane&15
__device__ __forceinline__ v16bf load_frag_b(const bf16* __restrict__ p, int ld) {
  const int lane = threadIdx.x & 31;
  const int n = lane & 15;
  const int hh = lane >> 4;
  const bf16* base = p + (long)n * ld + hh * 8;
  v16bf r;
  *reinterpret_cast<v8bf*>(&r)       = *reinterpret_cast<const v8bf*>(base);
  *(reinterpret_cast<v8bf*>(&r) + 1) = *reinterpret_cast<const v8bf*>(base + 16);
  return r;
}

// ------------------------------ small kernels ------------------------------

__global__ void cvt_kernel(const float* __restrict__ in, bf16* __restrict__ out, long n) {
  long i = (long)blockIdx.x * blockDim.x + threadIdx.x;
  long stride = (long)gridDim.x * blockDim.x;
  for (; i < n; i += stride) out[i] = (bf16)in[i];
}

// out[c][r] = in[r][c] * scale, per batch z (rows,cols multiples of 32)
__global__ __launch_bounds__(256)
void transpose_cvt_kernel(const float* __restrict__ in, bf16* __restrict__ out,
                          int rows, int cols, float scale) {
  __shared__ float t[32][33];
  const long zoff = (long)blockIdx.z * rows * cols;
  const int c0 = blockIdx.x * 32, r0 = blockIdx.y * 32;
  const int tx = threadIdx.x & 31, ty = threadIdx.x >> 5;
#pragma unroll
  for (int i = ty; i < 32; i += 8)
    t[i][tx] = in[zoff + (long)(r0 + i) * cols + c0 + tx];
  __syncthreads();
#pragma unroll
  for (int i = ty; i < 32; i += 8)
    out[zoff + (long)(c0 + i) * rows + r0 + tx] = (bf16)(t[tx][i] * scale);
}

// Tt[kh][r][j] = w[kh, j - r + 127]  (column-major Toeplitz for conv-as-GEMM)
__global__ void toeplitz_kernel(const float* __restrict__ w, bf16* __restrict__ Tt) {
  int idx = blockIdx.x * 256 + threadIdx.x;     // < 3*256*256
  int kh = idx >> 16;
  int r  = (idx >> 8) & 255;
  int j  = idx & 255;
  int kw = j - r + 127;
  float v = (kw >= 0 && kw < 256) ? w[kh * 256 + kw] : 0.f;
  Tt[idx] = (bf16)v;
}

// ------------------------------ generic GEMM -------------------------------
// C[M,N] = A[M,K] * Bt[N,K]^T ; batched via z = zb*hmod + zh with per-part strides.
// Double-buffered LDS tiles filled by async global->LDS copies.

template <int BM, int BN, int BK, int WR, int WC, bool TSTORE, bool OUTF32, bool BIAS>
__global__ __launch_bounds__(WR * WC * 32)
void gemm_kernel(const bf16* __restrict__ A, int lda,
                 const bf16* __restrict__ Bt, int ldb,
                 void* __restrict__ Cout, int ldc,
                 const float* __restrict__ bias, int K, int hmod,
                 long sAb, long sAh, long sBb, long sBh, long sCb, long sCh) {
  constexpr int BKP = BK + 8;           // pad: row stride stays 16B aligned (80B)
  __shared__ bf16 As[2][BM][BKP];
  __shared__ bf16 Bs[2][BN][BKP];

  const int z = blockIdx.z;
  const int zb = z / hmod, zh = z % hmod;
  A  += (long)zb * sAb + (long)zh * sAh + (long)blockIdx.y * BM * lda;
  Bt += (long)zb * sBb + (long)zh * sBh + (long)blockIdx.x * BN * ldb;
  const long cOff = (long)zb * sCb + (long)zh * sCh;

  const int tid = threadIdx.x;
  const int wave = tid >> 5, lane = tid & 31;
  const int wr = wave / WC, wc = wave % WC;
  constexpr int WM = BM / WR, WN = BN / WC;
  constexpr int FM = WM / 16, FN = WN / 16;
  constexpr int NT = WR * WC * 32;

  v8f acc[FM][FN];
#pragma unroll
  for (int fm = 0; fm < FM; ++fm)
#pragma unroll
    for (int fn = 0; fn < FN; ++fn)
#pragma unroll
      for (int i = 0; i < 8; ++i) acc[fm][fn][i] = 0.f;

  auto issue_tile = [&](int buf, int k0) {
    constexpr int AV = BM * (BK / 8);
#pragma unroll 2
    for (int idx = tid; idx < AV; idx += NT) {
      int r = idx / (BK / 8), c = (idx % (BK / 8)) * 8;
      async_ld_b128(&As[buf][r][c], A + (long)r * lda + k0 + c);
    }
    constexpr int BV = BN * (BK / 8);
#pragma unroll 2
    for (int idx = tid; idx < BV; idx += NT) {
      int r = idx / (BK / 8), c = (idx % (BK / 8)) * 8;
      async_ld_b128(&Bs[buf][r][c], Bt + (long)r * ldb + k0 + c);
    }
  };

  issue_tile(0, 0);
  const int nsteps = K / BK;
  for (int t = 0; t < nsteps; ++t) {
    wait_async0();          // this wave's async writes for buffer t&1 landed
    __syncthreads();        // all waves' writes landed; prev-tile readers done
    if (t + 1 < nsteps) issue_tile((t + 1) & 1, (t + 1) * BK);
    const int cur = t & 1;
#pragma unroll
    for (int kk = 0; kk < BK; kk += 32) {
      v16bf af[FM], bfv[FN];
#pragma unroll
      for (int fm = 0; fm < FM; ++fm)
        af[fm] = load_frag_a(&As[cur][wr * WM + fm * 16][kk], BKP);
#pragma unroll
      for (int fn = 0; fn < FN; ++fn)
        bfv[fn] = load_frag_b(&Bs[cur][wc * WN + fn * 16][kk], BKP);
#pragma unroll
      for (int fm = 0; fm < FM; ++fm)
#pragma unroll
        for (int fn = 0; fn < FN; ++fn)
          acc[fm][fn] = wmma_bf16(af[fm], bfv[fn], acc[fm][fn]);
    }
  }

  const int n16 = lane & 15, rb = (lane >> 4) * 8;
#pragma unroll
  for (int fm = 0; fm < FM; ++fm)
#pragma unroll
    for (int fn = 0; fn < FN; ++fn) {
      const int gm = blockIdx.y * BM + wr * WM + fm * 16 + rb;
      const int gn = blockIdx.x * BN + wc * WN + fn * 16 + n16;
#pragma unroll
      for (int i = 0; i < 8; ++i) {
        float v = acc[fm][fn][i];
        if (BIAS) v += bias[gn];
        long o = TSTORE ? (cOff + (long)gn * ldc + gm + i)
                        : (cOff + (long)(gm + i) * ldc + gn);
        if (OUTF32) reinterpret_cast<float*>(Cout)[o] = v;
        else        reinterpret_cast<bf16*>(Cout)[o]  = (bf16)v;
      }
    }
}

// --------------- fused conv(Toeplitz GEMM) + softmax + attn@vp -------------
// grid: (S/128, B*H). block: 256 (8 waves). Each wave owns 16 output rows.

__global__ __launch_bounds__(256)
void conv_softmax_av_kernel(const bf16* __restrict__ scores,
                            const bf16* __restrict__ vpT,   // [B*H][64][256]
                            const bf16* __restrict__ Tt,    // [3][256][256]
                            bf16* __restrict__ attnout) {   // [16384][1024]
  extern __shared__ __align__(16) bf16 smem[];
  bf16* sc = smem;                 // 130 x 256 (rows s0-1 .. s0+128, halo for conv)
  bf16* at = smem + 130 * 256;     // 128 x 256

  const int z = blockIdx.y;        // b*H + h
  const int b = z >> 4, h = z & 15;
  const int s0 = blockIdx.x * 128;
  const bf16* scb = scores + (long)z * Ss * Rr;

  if (s0 >= 1 && s0 + 129 <= Ss) {
    // interior tile: pure async global->LDS staging
    for (int idx = threadIdx.x; idx < 130 * 32; idx += 256) {
      int r = idx >> 5, c = (idx & 31) * 8;
      async_ld_b128(&sc[r * 256 + c], scb + (long)(s0 - 1 + r) * Rr + c);
    }
    wait_async0();
  } else {
    v8bf zv;
#pragma unroll
    for (int t = 0; t < 8; ++t) zv[t] = (bf16)0.f;
    for (int idx = threadIdx.x; idx < 130 * 32; idx += 256) {
      int r = idx >> 5, c = (idx & 31) * 8;
      int s = s0 - 1 + r;
      v8bf val = zv;
      if (s >= 0 && s < Ss)
        val = *reinterpret_cast<const v8bf*>(scb + (long)s * Rr + c);
      *reinterpret_cast<v8bf*>(&sc[r * 256 + c]) = val;
    }
  }
  __syncthreads();

  const int wave = threadIdx.x >> 5, lane = threadIdx.x & 31;
  const int m0 = wave * 16;        // local output row base

  v8f acc[16];
#pragma unroll
  for (int f = 0; f < 16; ++f)
#pragma unroll
    for (int i = 0; i < 8; ++i) acc[f][i] = 0.f;

  // conv = sum_kh shift_kh(scores) @ T_kh   (K = 256)
  for (int k0 = 0; k0 < 256; k0 += 32) {
    v16bf a[3];
#pragma unroll
    for (int kh = 0; kh < 3; ++kh)
      a[kh] = load_frag_a(sc + (m0 + kh) * 256 + k0, 256);
#pragma unroll
    for (int f = 0; f < 16; ++f) {
#pragma unroll
      for (int kh = 0; kh < 3; ++kh) {
        v16bf bt = load_frag_b(Tt + kh * 65536 + f * 16 * 256 + k0, 256);
        acc[f] = wmma_bf16(a[kh], bt, acc[f]);
      }
    }
  }

  // row-wise softmax over 256 cols. C-frag: VGPR i -> row i (lanes 0-15),
  // row 8+i (lanes 16-31); xor masks 1,2,4,8 stay within each 16-lane half.
  float mx[8], sm[8];
#pragma unroll
  for (int i = 0; i < 8; ++i) { mx[i] = -3.4e38f; sm[i] = 0.f; }
#pragma unroll
  for (int f = 0; f < 16; ++f)
#pragma unroll
    for (int i = 0; i < 8; ++i) mx[i] = fmaxf(mx[i], acc[f][i]);
#pragma unroll
  for (int d = 1; d < 16; d <<= 1)
#pragma unroll
    for (int i = 0; i < 8; ++i) mx[i] = fmaxf(mx[i], __shfl_xor(mx[i], d, 32));
#pragma unroll
  for (int f = 0; f < 16; ++f)
#pragma unroll
    for (int i = 0; i < 8; ++i) {
      float e = __expf(acc[f][i] - mx[i]);
      acc[f][i] = e;
      sm[i] += e;
    }
#pragma unroll
  for (int d = 1; d < 16; d <<= 1)
#pragma unroll
    for (int i = 0; i < 8; ++i) sm[i] += __shfl_xor(sm[i], d, 32);
  float inv[8];
#pragma unroll
  for (int i = 0; i < 8; ++i) inv[i] = 1.f / sm[i];

  const int nn = lane & 15, rb = (lane >> 4) * 8;
#pragma unroll
  for (int f = 0; f < 16; ++f)
#pragma unroll
    for (int i = 0; i < 8; ++i)
      at[(m0 + rb + i) * 256 + f * 16 + nn] = (bf16)(acc[f][i] * inv[i]);
  __syncthreads();

  // out = attn[128,256] @ vp[256,64]  (vpT is column-major B: [64][256])
  const bf16* vb = vpT + (long)z * 64 * 256;
  v8f o[4];
#pragma unroll
  for (int f = 0; f < 4; ++f)
#pragma unroll
    for (int i = 0; i < 8; ++i) o[f][i] = 0.f;
  for (int k0 = 0; k0 < 256; k0 += 32) {
    v16bf a = load_frag_a(at + m0 * 256 + k0, 256);
#pragma unroll
    for (int f = 0; f < 4; ++f) {
      v16bf bt = load_frag_b(vb + f * 16 * 256 + k0, 256);
      o[f] = wmma_bf16(a, bt, o[f]);
    }
  }
  const long rowb = (long)(b * Ss + s0 + m0 + rb);
#pragma unroll
  for (int f = 0; f < 4; ++f)
#pragma unroll
    for (int i = 0; i < 8; ++i)
      attnout[(rowb + i) * Dd + h * 64 + f * 16 + nn] = (bf16)o[f][i];
}

// ------------------------------ host launcher ------------------------------

extern "C" void kernel_launch(void* const* d_in, const int* in_sizes, int n_in,
                              void* d_out, int out_size, void* d_ws, size_t ws_size,
                              hipStream_t stream) {
  (void)in_sizes; (void)n_in; (void)out_size; (void)ws_size;
  const float* x   = (const float*)d_in[0];
  const float* wq  = (const float*)d_in[1];
  const float* wk  = (const float*)d_in[2];
  const float* wv  = (const float*)d_in[3];
  const float* E   = (const float*)d_in[4];
  const float* F   = (const float*)d_in[5];
  const float* ck  = (const float*)d_in[6];
  // d_in[7] = conv_bias: uniform over softmax axis -> no effect, skipped.
  const float* wo  = (const float*)d_in[8];
  const float* bo  = (const float*)d_in[9];

  char* ws = (char*)d_ws;
  // workspace layout (bytes), lifetime-aliased; total ~205 MB
  const long sz_x = NTOK * Dd * 2;            // 33.5 MB per [16384,1024] bf16
  const long sz_w = (long)Dd * Dd * 2;        // 2 MB
  const long o_x  = 0;                        // xbf, later reused as attnout
  const long o_wq = o_x + sz_x;
  const long o_wk = o_wq + sz_w;
  const long o_wv = o_wk + sz_w;
  const long o_wo = o_wv + sz_w;
  const long o_q  = o_wo + sz_w;
  const long o_kt = o_q + sz_x;               // kT [1024,16384]
  const long o_vt = o_kt + sz_x;              // vT [1024,16384]
  const long o_Et = o_vt + sz_x;              // E^T*scale [16][256][4096]
  const long o_Ft = o_Et + sz_x;              // F^T       [16][256][4096]
  const long o_scores = o_kt;                 // reuses kt..Ft (exactly 134 MB)
  const long o_kp  = o_Ft + sz_x;             // kp  [64][256][64]
  const long o_vpt = o_kp + 2097152;          // vpT [64][64][256]
  const long o_tt  = o_vpt + 2097152;         // Toeplitz [3][256][256]

  bf16* xbf = (bf16*)(ws + o_x);
  bf16* wqt = (bf16*)(ws + o_wq);
  bf16* wkt = (bf16*)(ws + o_wk);
  bf16* wvt = (bf16*)(ws + o_wv);
  bf16* wot = (bf16*)(ws + o_wo);
  bf16* qb  = (bf16*)(ws + o_q);
  bf16* kt2 = (bf16*)(ws + o_kt);
  bf16* vt2 = (bf16*)(ws + o_vt);
  bf16* Etb = (bf16*)(ws + o_Et);
  bf16* Ftb = (bf16*)(ws + o_Ft);
  bf16* scb = (bf16*)(ws + o_scores);
  bf16* kp  = (bf16*)(ws + o_kp);
  bf16* vpt = (bf16*)(ws + o_vpt);
  bf16* tt  = (bf16*)(ws + o_tt);
  bf16* attnout = (bf16*)(ws + o_x);          // alias of xbf (dead after QKV)

  const float scale = 0.125f;                 // 1/sqrt(64), folded into E^T

  // 1) conversions / transposes / Toeplitz build
  cvt_kernel<<<8192, 256, 0, stream>>>(x, xbf, NTOK * Dd);
  transpose_cvt_kernel<<<dim3(32, 32, 1), 256, 0, stream>>>(wq, wqt, Dd, Dd, 1.f);
  transpose_cvt_kernel<<<dim3(32, 32, 1), 256, 0, stream>>>(wk, wkt, Dd, Dd, 1.f);
  transpose_cvt_kernel<<<dim3(32, 32, 1), 256, 0, stream>>>(wv, wvt, Dd, Dd, 1.f);
  transpose_cvt_kernel<<<dim3(32, 32, 1), 256, 0, stream>>>(wo, wot, Dd, Dd, 1.f);
  transpose_cvt_kernel<<<dim3(8, 128, 16), 256, 0, stream>>>(E, Etb, Ss, Rr, scale);
  transpose_cvt_kernel<<<dim3(8, 128, 16), 256, 0, stream>>>(F, Ftb, Ss, Rr, 1.f);
  toeplitz_kernel<<<768, 256, 0, stream>>>(ck, tt);

  // 2) QKV: [16384,1024] @ w^T ; k,v stored transposed ([D, B*S]) for proj stage
  gemm_kernel<128, 128, 32, 2, 4, false, false, false>
      <<<dim3(8, 128, 1), 256, 0, stream>>>(xbf, Dd, wqt, Dd, qb, Dd, nullptr,
                                            Dd, 1, 0, 0, 0, 0, 0, 0);
  gemm_kernel<128, 128, 32, 2, 4, true, false, false>
      <<<dim3(8, 128, 1), 256, 0, stream>>>(xbf, Dd, wkt, Dd, kt2, (int)NTOK, nullptr,
                                            Dd, 1, 0, 0, 0, 0, 0, 0);
  gemm_kernel<128, 128, 32, 2, 4, true, false, false>
      <<<dim3(8, 128, 1), 256, 0, stream>>>(xbf, Dd, wvt, Dd, vt2, (int)NTOK, nullptr,
                                            Dd, 1, 0, 0, 0, 0, 0, 0);

  // 3) low-rank projections per (b,h): kp = (E^T*scale) @ k_head ; vpT = (F^T @ v_head)^T
  gemm_kernel<128, 64, 32, 4, 2, false, false, false>
      <<<dim3(1, 2, Bb * Hh), 256, 0, stream>>>(
          Etb, Ss, kt2, (int)NTOK, kp, DEPTH, nullptr, Ss, Hh,
          0, (long)Rr * Ss, (long)Ss, (long)DEPTH * NTOK,
          (long)Hh * Rr * DEPTH, (long)Rr * DEPTH);
  gemm_kernel<128, 64, 32, 4, 2, true, false, false>
      <<<dim3(1, 2, Bb * Hh), 256, 0, stream>>>(
          Ftb, Ss, vt2, (int)NTOK, vpt, Rr, nullptr, Ss, Hh,
          0, (long)Rr * Ss, (long)Ss, (long)DEPTH * NTOK,
          (long)Hh * Rr * DEPTH, (long)Rr * DEPTH);

  // 4) scores[b,h] = q_head @ kp^T  (K = 64), bf16 out
  gemm_kernel<128, 128, 32, 2, 4, false, false, false>
      <<<dim3(2, 32, Bb * Hh), 256, 0, stream>>>(
          qb, Dd, kp, DEPTH, scb, Rr, nullptr, DEPTH, Hh,
          (long)Ss * Dd, (long)DEPTH, (long)Hh * Rr * DEPTH, (long)Rr * DEPTH,
          (long)Hh * Ss * Rr, (long)Ss * Rr);

  // 5) fused Toeplitz-conv + softmax + attn@v_proj  (132 KB dynamic LDS)
  conv_softmax_av_kernel<<<dim3(Ss / 128, Bb * Hh), 256,
                           (130 * 256 + 128 * 256) * (int)sizeof(bf16), stream>>>(
      scb, vpt, tt, attnout);

  // 6) final: out = attnout @ wo + bo  (f32 out)
  gemm_kernel<128, 128, 32, 2, 4, false, true, true>
      <<<dim3(8, 128, 1), 256, 0, stream>>>(attnout, Dd, wot, Dd, (float*)d_out,
                                            Dd, bo, Dd, 1, 0, 0, 0, 0, 0, 0);
}
